// EncoderLayer_19344532701215
// MI455X (gfx1250) — matrix operand order repaired and checked by reading, not compiled, not run on hardware
//
#include <hip/hip_runtime.h>
#include <hip/hip_bf16.h>
#include <math.h>

// ---------------------------------------------------------------------------
// Types for CDNA5 WMMA (gfx1250, wave32)
// ---------------------------------------------------------------------------
typedef __attribute__((ext_vector_type(16))) __bf16 v16bf;
typedef __attribute__((ext_vector_type(8)))  float  v8f;
typedef __attribute__((ext_vector_type(4)))  unsigned int v4u;
typedef __attribute__((ext_vector_type(2)))  float  v2f;

#define D_MODEL 512
#define SEQ     256
#define BATCH   2
#define TOKENS  (BATCH * SEQ)   // 512
#define HEADS   8
#define D_K     64
#define DIM_FFN 2048

__device__ __forceinline__ unsigned short f2bf(float f) {
    unsigned int u = __float_as_uint(f);
    unsigned int r = u + 0x7FFFu + ((u >> 16) & 1u);   // round-to-nearest-even
    return (unsigned short)(r >> 16);
}

__device__ __forceinline__ float wave_sum(float v) {
    for (int m = 16; m >= 1; m >>= 1) v += __shfl_xor(v, m, 32);
    return v;
}
__device__ __forceinline__ float wave_max(float v) {
    for (int m = 16; m >= 1; m >>= 1) v = fmaxf(v, __shfl_xor(v, m, 32));
    return v;
}

// ---------------------------------------------------------------------------
// fp32 -> bf16 bulk convert
// ---------------------------------------------------------------------------
__global__ void __launch_bounds__(256)
cvt_bf16_kernel(const float* __restrict__ in, unsigned short* __restrict__ out, int n) {
    int i = blockIdx.x * 256 + threadIdx.x;
    if (i < n) out[i] = f2bf(in[i]);
}

// ---------------------------------------------------------------------------
// LayerNorm (ddof=1, alpha*(x-mu)/(sd+eps)+beta), wave per row, bf16 output
// ---------------------------------------------------------------------------
__global__ void __launch_bounds__(256)
layernorm_bf16_kernel(const float* __restrict__ X,
                      const float* __restrict__ alpha,
                      const float* __restrict__ beta,
                      unsigned short* __restrict__ Y, int nrows) {
    int wave = threadIdx.x >> 5, lane = threadIdx.x & 31;
    int row = blockIdx.x * 8 + wave;
    if (row >= nrows) return;
    const float* xr = X + (size_t)row * D_MODEL;
    float vals[16];
    float s = 0.f, ss = 0.f;
    for (int t = 0; t < 16; ++t) {
        float v = xr[lane + 32 * t];
        vals[t] = v; s += v; ss += v * v;
    }
    s  = wave_sum(s);
    ss = wave_sum(ss);
    float mu  = s * (1.f / 512.f);
    float var = fmaxf((ss - 512.f * mu * mu) * (1.f / 511.f), 0.f);
    float inv = 1.f / (sqrtf(var) + 1e-6f);
    for (int t = 0; t < 16; ++t) {
        int d = lane + 32 * t;
        float y = alpha[d] * (vals[t] - mu) * inv + beta[d];
        Y[(size_t)row * D_MODEL + d] = f2bf(y);
    }
}

// ---------------------------------------------------------------------------
// GEMM: C[M,N] = act(A[M,K] @ W[N,K]^T + bias (+ residual))
// A,W bf16 row-major (contiguous K).  One wave -> one 16x32 C strip
// (two 16x16 subtiles; A fragment reused across both WMMAs).
// Fragment layouts as loaded here (A/B contiguous K; C per CDNA5 ISA 7.12.2):
//   A 16x32: lanes 0-15 row M=lane,  K in {0..15};
//            lanes 16-31 row M=lane-16, K in {16..31}
//   B 32x16: lanes 0-15 col N=lane,  K in {0..15};
//            lanes 16-31 col N=lane-16, K in {16..31}
//   C 16x16: vgpr r, lanes 0-15 -> (M=r,   N=lane)
//            vgpr r, lanes16-31 -> (M=8+r, N=lane-16)
// ---------------------------------------------------------------------------
__global__ void __launch_bounds__(256)
gemm_bf16_wmma_kernel(const unsigned short* __restrict__ A,
                      const unsigned short* __restrict__ W,
                      const float* __restrict__ bias,
                      const float* __restrict__ residual,
                      float* __restrict__ Cf,
                      unsigned short* __restrict__ Cbf,
                      int M, int N, int K, int relu) {
    int lane = threadIdx.x & 31;
    int wid  = (blockIdx.x * 256 + threadIdx.x) >> 5;
    int ngrp  = N >> 5;                 // 32-wide column groups
    int tiles = (M >> 4) * ngrp;
    if (wid >= tiles) return;
    int m0 = (wid / ngrp) << 4;
    int n0 = (wid % ngrp) << 5;

    int hi = (lane & 16) ? 1 : 0;
    const unsigned short* ap  = A + (size_t)(m0 + (lane & 15)) * K + (hi ? 16 : 0);
    const unsigned short* wp0 = W + (size_t)(n0 + (lane & 15)) * K + (hi ? 16 : 0);
    const unsigned short* wp1 = wp0 + (size_t)16 * K;

    v8f acc0 = {}, acc1 = {};
    for (int k0 = 0; k0 < K; k0 += 32) {
        union { v16bf v; v4u q[2]; } fa, fb0, fb1;
        fa.q[0]  = *(const v4u*)(ap + k0);         // K = koff + 0..7
        fa.q[1]  = *(const v4u*)(ap + k0 + 8);     // K = koff + 8..15
        fb0.q[0] = *(const v4u*)(wp0 + k0);        // K = kbase + 0..7
        fb0.q[1] = *(const v4u*)(wp0 + k0 + 8);    // K = kbase + 8..15
        fb1.q[0] = *(const v4u*)(wp1 + k0);
        fb1.q[1] = *(const v4u*)(wp1 + k0 + 8);
        acc0 = __builtin_amdgcn_wmma_f32_16x16x32_bf16(
            false, fa.v, false, fb0.v, (short)0, acc0, false, false);
        acc1 = __builtin_amdgcn_wmma_f32_16x16x32_bf16(
            false, fa.v, false, fb1.v, (short)0, acc1, false, false);
    }

    int mBase = m0 + (hi ? 8 : 0);
    for (int sub = 0; sub < 2; ++sub) {
        v8f acc = sub ? acc1 : acc0;
        int nOut = n0 + sub * 16 + (lane & 15);
        float bv = bias ? bias[nOut] : 0.f;
        for (int r = 0; r < 8; ++r) {
            float c = acc[r] + bv;
            size_t idx = (size_t)(mBase + r) * N + nOut;
            if (residual) c += residual[idx];
            if (relu) c = fmaxf(c, 0.f);
            if (Cf)  Cf[idx]  = c;
            if (Cbf) Cbf[idx] = f2bf(c);
        }
    }
}

// ---------------------------------------------------------------------------
// Fused attention: scores(+bias_k) -> softmax -> attn@(v+bias_v).
// One wave per (b,h,i); attention row lives in wave-private LDS.
//   scores[j] = (1/8) * sum_d q[t,c+d] * (k[tj,c+d] + cbk[b,j,i,c+d])
//   out[t,c+d] = sum_j attn[j] * (v[tj,c+d] + cbv[b,j,i,c+d])
// Lanes span d (float2 each) -> coalesced 256B per j.  cbk/cbv are streamed
// once -> non-temporal loads keep them out of the caches that hold k/v.
// Masked query rows (mask[b,i]==1) -> uniform attention 1/256.
// ---------------------------------------------------------------------------
__global__ void __launch_bounds__(256)
attn_fused_kernel(const float* __restrict__ q,
                  const float* __restrict__ k,
                  const float* __restrict__ v,
                  const float* __restrict__ cbk,
                  const float* __restrict__ cbv,
                  const int* __restrict__ mask,
                  unsigned short* __restrict__ concat_bf) {
    __shared__ float srow[8][SEQ];
    int wave = threadIdx.x >> 5, lane = threadIdx.x & 31;
    int gid = blockIdx.x * 8 + wave;          // (b*8 + h)*256 + i
    int i  = gid & 255;
    int hh = (gid >> 8) & 7;
    int bb = gid >> 11;
    int c  = hh * D_K + 2 * lane;
    int t  = bb * SEQ + i;

    // ---- phase 1: scores + softmax ----
    v2f qv = *(const v2f*)(q + (size_t)t * D_MODEL + c);
    const float* kb  = k + (size_t)bb * SEQ * D_MODEL + c;
    const float* cbA = cbk + ((size_t)bb * SEQ * SEQ + i) * D_MODEL + c;

    for (int j = 0; j < SEQ; ++j) {
        v2f kv = *(const v2f*)(kb + (size_t)j * D_MODEL);
        v2f bk = __builtin_nontemporal_load((const v2f*)(cbA + (size_t)j * SEQ * D_MODEL));
        float p = qv.x * (kv.x + bk.x) + qv.y * (kv.y + bk.y);
        p = wave_sum(p);
        if (lane == 0) srow[wave][j] = p * 0.125f;   // 1/sqrt(64)
    }
    // wave-private LDS row: same-wave DS ops are in-order, no barrier needed.
    bool masked = (mask[bb * SEQ + i] == 1);
    float vals[8];
    float mx = -3.0e38f;
    for (int tt = 0; tt < 8; ++tt) {
        vals[tt] = srow[wave][lane + 32 * tt];
        mx = fmaxf(mx, vals[tt]);
    }
    mx = wave_max(mx);
    float sum = 0.f;
    for (int tt = 0; tt < 8; ++tt) {
        float e = __expf(vals[tt] - mx);
        vals[tt] = e; sum += e;
    }
    sum = wave_sum(sum);
    float inv = 1.f / sum;
    for (int tt = 0; tt < 8; ++tt)
        srow[wave][lane + 32 * tt] = masked ? (1.f / SEQ) : vals[tt] * inv;

    // ---- phase 2: output accumulation ----
    const float* vb  = v + (size_t)bb * SEQ * D_MODEL + c;
    const float* cbB = cbv + ((size_t)bb * SEQ * SEQ + i) * D_MODEL + c;
    float ax = 0.f, ay = 0.f;
    for (int j = 0; j < SEQ; ++j) {
        float a = srow[wave][j];                     // LDS broadcast
        v2f vv = *(const v2f*)(vb + (size_t)j * D_MODEL);
        v2f bv = __builtin_nontemporal_load((const v2f*)(cbB + (size_t)j * SEQ * D_MODEL));
        ax = fmaf(a, vv.x + bv.x, ax);
        ay = fmaf(a, vv.y + bv.y, ay);
    }
    concat_bf[(size_t)t * D_MODEL + c]     = f2bf(ax);
    concat_bf[(size_t)t * D_MODEL + c + 1] = f2bf(ay);
}

// ---------------------------------------------------------------------------
// Host-side launcher
// ---------------------------------------------------------------------------
extern "C" void kernel_launch(void* const* d_in, const int* in_sizes, int n_in,
                              void* d_out, int out_size, void* d_ws, size_t ws_size,
                              hipStream_t stream) {
    (void)in_sizes; (void)n_in; (void)out_size; (void)ws_size;
    const float* x      = (const float*)d_in[0];
    const float* cbk    = (const float*)d_in[1];
    const float* cbv    = (const float*)d_in[2];
    const int*   mask   = (const int*)d_in[3];
    const float* Wq     = (const float*)d_in[4];
    const float* bq     = (const float*)d_in[5];
    const float* Wk     = (const float*)d_in[6];
    const float* bk     = (const float*)d_in[7];
    const float* Wv     = (const float*)d_in[8];
    const float* bv     = (const float*)d_in[9];
    const float* Wo     = (const float*)d_in[10];
    const float* bo     = (const float*)d_in[11];
    const float* alpha1 = (const float*)d_in[12];
    const float* beta1  = (const float*)d_in[13];
    const float* alpha2 = (const float*)d_in[14];
    const float* beta2  = (const float*)d_in[15];
    const float* W1     = (const float*)d_in[16];
    const float* b1     = (const float*)d_in[17];
    const float* W2     = (const float*)d_in[18];
    const float* b2     = (const float*)d_in[19];
    float* out = (float*)d_out;

    // ---- workspace carve-up (256B aligned) ----
    char* base = (char*)d_ws;
    size_t off = 0;
    auto alloc = [&](size_t bytes) -> void* {
        void* p = base + off;
        off += (bytes + 255) & ~(size_t)255;
        return p;
    };
    unsigned short* wq_bf = (unsigned short*)alloc(512 * 512 * 2);
    unsigned short* wk_bf = (unsigned short*)alloc(512 * 512 * 2);
    unsigned short* wv_bf = (unsigned short*)alloc(512 * 512 * 2);
    unsigned short* wo_bf = (unsigned short*)alloc(512 * 512 * 2);
    unsigned short* w1_bf = (unsigned short*)alloc(2048 * 512 * 2);
    unsigned short* w2_bf = (unsigned short*)alloc(512 * 2048 * 2);
    unsigned short* x2a_bf = (unsigned short*)alloc(TOKENS * 512 * 2);
    float* qf = (float*)alloc(TOKENS * 512 * 4);
    float* kf = (float*)alloc(TOKENS * 512 * 4);
    float* vf = (float*)alloc(TOKENS * 512 * 4);
    unsigned short* concat_bf = (unsigned short*)alloc(TOKENS * 512 * 2);
    float* xres = (float*)alloc(TOKENS * 512 * 4);
    unsigned short* x2b_bf = (unsigned short*)alloc(TOKENS * 512 * 2);
    unsigned short* ff1_bf = (unsigned short*)alloc((size_t)TOKENS * 2048 * 2);

    // ---- 1) weight conversion fp32 -> bf16 ----
    cvt_bf16_kernel<<<(512 * 512) / 256, 256, 0, stream>>>(Wq, wq_bf, 512 * 512);
    cvt_bf16_kernel<<<(512 * 512) / 256, 256, 0, stream>>>(Wk, wk_bf, 512 * 512);
    cvt_bf16_kernel<<<(512 * 512) / 256, 256, 0, stream>>>(Wv, wv_bf, 512 * 512);
    cvt_bf16_kernel<<<(512 * 512) / 256, 256, 0, stream>>>(Wo, wo_bf, 512 * 512);
    cvt_bf16_kernel<<<(2048 * 512) / 256, 256, 0, stream>>>(W1, w1_bf, 2048 * 512);
    cvt_bf16_kernel<<<(512 * 2048) / 256, 256, 0, stream>>>(W2, w2_bf, 512 * 2048);

    // ---- 2) LN1 -> bf16 activations ----
    layernorm_bf16_kernel<<<TOKENS / 8, 256, 0, stream>>>(x, alpha1, beta1, x2a_bf, TOKENS);

    // ---- 3) Q/K/V projections: 512x512x512 WMMA GEMMs ----
    {
        int blocks = ((TOKENS / 16) * (512 / 32)) / 8;   // 64
        gemm_bf16_wmma_kernel<<<blocks, 256, 0, stream>>>(x2a_bf, wq_bf, bq, nullptr, qf, nullptr, TOKENS, 512, 512, 0);
        gemm_bf16_wmma_kernel<<<blocks, 256, 0, stream>>>(x2a_bf, wk_bf, bk, nullptr, kf, nullptr, TOKENS, 512, 512, 0);
        gemm_bf16_wmma_kernel<<<blocks, 256, 0, stream>>>(x2a_bf, wv_bf, bv, nullptr, vf, nullptr, TOKENS, 512, 512, 0);
    }

    // ---- 4) fused scores+softmax+output : streams 512MB of cbk/cbv once ----
    attn_fused_kernel<<<(BATCH * HEADS * SEQ) / 8, 256, 0, stream>>>(
        qf, kf, vf, cbk, cbv, mask, concat_bf);

    // ---- 5) output projection + residual ----
    gemm_bf16_wmma_kernel<<<((TOKENS / 16) * (512 / 32)) / 8, 256, 0, stream>>>(
        concat_bf, wo_bf, bo, x, xres, nullptr, TOKENS, 512, 512, 0);

    // ---- 6) LN2 ----
    layernorm_bf16_kernel<<<TOKENS / 8, 256, 0, stream>>>(xres, alpha2, beta2, x2b_bf, TOKENS);

    // ---- 7) FFN1 (ReLU) -> bf16 only ----
    gemm_bf16_wmma_kernel<<<((TOKENS / 16) * (2048 / 32)) / 8, 256, 0, stream>>>(
        x2b_bf, w1_bf, b1, nullptr, nullptr, ff1_bf, TOKENS, 2048, 512, 1);

    // ---- 8) FFN2 + residual -> final output ----
    gemm_bf16_wmma_kernel<<<((TOKENS / 16) * (512 / 32)) / 8, 256, 0, stream>>>(
        ff1_bf, w2_bf, b2, xres, out, nullptr, TOKENS, 512, 2048, 0);
}
